// FluxSingleTransformerBlock_56994216018095
// MI455X (gfx1250) — compile-verified, matrix operand order checked by reading, not run on hardware
//
#include <hip/hip_runtime.h>
#include <math.h>

// ---------------- problem constants ----------------
#define DIMC      3072
#define SEQC      2048
#define HEADSC    24
#define HDC       128
#define MLPC      12288
#define QKVC      (3*DIMC)            // 9216
#define N1C       (QKVC + MLPC)       // 21504
#define CATC      (DIMC + MLPC)       // 15360
#define EPSC      1e-6f

typedef __attribute__((ext_vector_type(16))) __bf16 v16bf;
typedef __attribute__((ext_vector_type(8)))  float  v8f;
typedef __attribute__((ext_vector_type(4)))  unsigned int v4u;
typedef __attribute__((ext_vector_type(8)))  int    v8i;
typedef __attribute__((ext_vector_type(4)))  int    v4i;

// ---- TDM availability (device pass only; host pass takes fallback) ----
#if defined(__HIP_DEVICE_COMPILE__) && __has_builtin(__builtin_amdgcn_tensor_load_to_lds)
#define USE_TDM 1
#else
#define USE_TDM 0
#endif

// ---- fragment loaders (CDNA5 WMMA bf16 16x16x32 layout) ----
// per-lane K pattern: elems[0:8] at p, elems[8:16] at p+16 (p already includes
// the half-wave offset of +8 for lanes 16..31)
__device__ inline v16bf ldfrag_bf16(const __bf16* p) {
    union { v16bf v; uint4 u[2]; } r;
    r.u[0] = *(const uint4*)(p);
    r.u[1] = *(const uint4*)(p + 16);
    return r.v;
}

__device__ inline v16bf ldfrag_f32(const float* p) {
    float4 a = *(const float4*)(p);
    float4 b = *(const float4*)(p + 4);
    float4 c = *(const float4*)(p + 16);
    float4 d = *(const float4*)(p + 20);
    v16bf r;
    r[0]=(__bf16)a.x;  r[1]=(__bf16)a.y;  r[2]=(__bf16)a.z;  r[3]=(__bf16)a.w;
    r[4]=(__bf16)b.x;  r[5]=(__bf16)b.y;  r[6]=(__bf16)b.z;  r[7]=(__bf16)b.w;
    r[8]=(__bf16)c.x;  r[9]=(__bf16)c.y;  r[10]=(__bf16)c.z; r[11]=(__bf16)c.w;
    r[12]=(__bf16)d.x; r[13]=(__bf16)d.y; r[14]=(__bf16)d.z; r[15]=(__bf16)d.w;
    return r;
}

// weight-type dispatch for templated GEMMs
__device__ inline v16bf ldfrag_w(const __bf16* p) { return ldfrag_bf16(p); }
__device__ inline v16bf ldfrag_w(const float*  p) { return ldfrag_f32(p);  }

#define WMMA_BF16(A, B, C) \
    __builtin_amdgcn_wmma_f32_16x16x32_bf16(false, (A), false, (B), (short)0, (C), false, false)

__device__ inline float tanh_fast(float x) {
#if __has_builtin(__builtin_amdgcn_tanhf)
    return __builtin_amdgcn_tanhf(x);      // v_tanh_f32 (CDNA5 transcendental)
#else
    return tanhf(x);
#endif
}

__device__ inline float gelu_tanh(float x) {
    float x3 = x * x * x;
    return 0.5f * x * (1.0f + tanh_fast(0.7978845608028654f * (x + 0.044715f * x3)));
}

#if USE_TDM
// ---- Tensor Data Mover: DMA a 64x256B tile (64 keys x 128 bf16) into LDS ----
// D# per CDNA5 ISA ch.8: group0 = {count/type, lds_addr, global_addr lo/hi},
// group1 = {data_size, tensor dims, tile dims, strides}. data_size = 8B units:
// tile_dim0 = 32 (256B row), tile_dim1 = 64 rows, tensor_dim0_stride = 768
// (6144B = one K row of 3072 bf16). workgroup_mask = 0 (not in a cluster).
__device__ inline void tdm_load_k_tile(unsigned lds_off, unsigned long long gaddr) {
    v4u g0;
    g0[0] = 1u;                                            // count=1, user D#
    g0[1] = lds_off;                                       // lds_addr (bytes)
    g0[2] = (unsigned)(gaddr & 0xFFFFFFFFu);               // global_addr[31:0]
    g0[3] = ((unsigned)(gaddr >> 32) & 0x01FFFFFFu)        // global_addr[56:32]
            | (2u << 30);                                  // type = 2 ("image")
    v8i g1;
    g1[0] = (3 << 16);          // data_size = 3 (8 bytes); mask/flags = 0
    g1[1] = (32 << 16);         // tensor_dim0 = 32 units (low 16 bits)
    g1[2] = (64 << 16);         // tensor_dim0 hi = 0 ; tensor_dim1 = 64
    g1[3] = (32 << 16);         // tensor_dim1 hi = 0 ; tile_dim0 = 32
    g1[4] = 64;                 // tile_dim1 = 64 ; tile_dim2 = 0
    g1[5] = 768;                // tensor_dim0_stride low32 = 768 units
    g1[6] = 0;                  // stride hi ; tensor_dim1_stride lo
    g1[7] = 0;
    v4i z4 = {0, 0, 0, 0};
#if __clang_major__ >= 23
    v8i z8 = {0, 0, 0, 0, 0, 0, 0, 0};
    __builtin_amdgcn_tensor_load_to_lds(g0, g1, z4, z4, z8, 0);
#else
    __builtin_amdgcn_tensor_load_to_lds(g0, g1, z4, z4, 0);
#endif
}
#endif

// ============== weight fp32 -> bf16 conversion (8 elems/thread) ==============
__global__ __launch_bounds__(256) void cvt_bf16_kernel(
    const float* __restrict__ s, __bf16* __restrict__ d)
{
    size_t i = ((size_t)blockIdx.x * 256 + threadIdx.x) * 8;
    float4 a = *(const float4*)&s[i];
    float4 b = *(const float4*)&s[i + 4];
    union { uint4 u; __bf16 e[8]; } r;
    r.e[0]=(__bf16)a.x; r.e[1]=(__bf16)a.y; r.e[2]=(__bf16)a.z; r.e[3]=(__bf16)a.w;
    r.e[4]=(__bf16)b.x; r.e[5]=(__bf16)b.y; r.e[6]=(__bf16)b.z; r.e[7]=(__bf16)b.w;
    *(uint4*)&d[i] = r.u;
}

// ============== kernel 0: mod = silu(temb) @ norm_w^T + norm_b ==============
__global__ __launch_bounds__(256) void mod_gemv_kernel(
    const float* __restrict__ temb, const float* __restrict__ nw,
    const float* __restrict__ nb, float* __restrict__ mod)
{
    const int lane = threadIdx.x & 31;
    const int wid  = threadIdx.x >> 5;
    const int n = blockIdx.x * 8 + wid;              // 0..9215
    const float* wrow = nw + (size_t)n * DIMC;
    float acc = 0.0f;
    for (int i = lane * 4; i < DIMC; i += 128) {
        float4 t = *(const float4*)&temb[i];
        float4 w = *(const float4*)&wrow[i];
        float s0 = t.x / (1.0f + __expf(-t.x));
        float s1 = t.y / (1.0f + __expf(-t.y));
        float s2 = t.z / (1.0f + __expf(-t.z));
        float s3 = t.w / (1.0f + __expf(-t.w));
        acc += s0 * w.x + s1 * w.y + s2 * w.z + s3 * w.w;
    }
    #pragma unroll
    for (int off = 16; off > 0; off >>= 1) acc += __shfl_xor(acc, off, 32);
    if (lane == 0) mod[n] = acc + nb[n];
}

// ============== kernel 1: LayerNorm + AdaLN modulate -> bf16 ==============
__global__ __launch_bounds__(256) void ln_kernel(
    const float* __restrict__ hid, const float* __restrict__ mod,
    __bf16* __restrict__ xn)
{
    const int s = blockIdx.x;
    const int t = threadIdx.x;
    const int lane = t & 31, wid = t >> 5;
    const float* row = hid + (size_t)s * DIMC;
    float4 v[3];
    float sum = 0.0f, sq = 0.0f;
    #pragma unroll
    for (int j = 0; j < 3; ++j) {
        v[j] = *(const float4*)&row[(t + 256 * j) * 4];
        sum += v[j].x + v[j].y + v[j].z + v[j].w;
        sq  += v[j].x*v[j].x + v[j].y*v[j].y + v[j].z*v[j].z + v[j].w*v[j].w;
    }
    #pragma unroll
    for (int off = 16; off > 0; off >>= 1) {
        sum += __shfl_xor(sum, off, 32);
        sq  += __shfl_xor(sq,  off, 32);
    }
    __shared__ float red[16];
    if (lane == 0) { red[wid] = sum; red[8 + wid] = sq; }
    __syncthreads();
    float tsum = 0.0f, tsq = 0.0f;
    #pragma unroll
    for (int i = 0; i < 8; ++i) { tsum += red[i]; tsq += red[8 + i]; }
    const float mu  = tsum * (1.0f / DIMC);
    const float var = tsq * (1.0f / DIMC) - mu * mu;
    const float rr  = rsqrtf(var + EPSC);
    const float* shift = mod;
    const float* scale = mod + DIMC;
    __bf16* orow = xn + (size_t)s * DIMC;
    #pragma unroll
    for (int j = 0; j < 3; ++j) {
        int base = (t + 256 * j) * 4;
        float x[4] = { v[j].x, v[j].y, v[j].z, v[j].w };
        #pragma unroll
        for (int c = 0; c < 4; ++c) {
            float y = (x[c] - mu) * rr * (1.0f + scale[base + c]) + shift[base + c];
            orow[base + c] = (__bf16)y;
        }
    }
}

// ============== kernel 2: GEMM1  xn @ qkv_mlp_w^T  (+bias, gelu split) ==============
// block 256 thr = 8 waves, tile M=128 N=128; wave tile 32x64 (2x4 wmma accs)
template <typename WT>
__global__ __launch_bounds__(256) void gemm1_kernel(
    const __bf16* __restrict__ xn, const WT* __restrict__ w,
    const float* __restrict__ bias, __bf16* __restrict__ qkv,
    __bf16* __restrict__ mlp)
{
    const int lane = threadIdx.x & 31;
    const int wid  = threadIdx.x >> 5;
    const int wm = wid & 3, wn = wid >> 2;
    const int m0 = blockIdx.y * 128 + wm * 32;
    const int n0 = blockIdx.x * 128 + wn * 64;
    const int lh = lane & 15;
    const int hoff = (lane & 16) ? 8 : 0;
    const bool is_qkv = (int)(blockIdx.x * 128) < QKVC;   // 9216 % 128 == 0

    const __bf16* a0 = xn + (size_t)(m0 + lh) * DIMC + hoff;
    const __bf16* a1 = xn + (size_t)(m0 + 16 + lh) * DIMC + hoff;
    const WT* b0 = w + (size_t)(n0 + lh) * DIMC + hoff;
    const WT* b1 = b0 + (size_t)16 * DIMC;
    const WT* b2 = b0 + (size_t)32 * DIMC;
    const WT* b3 = b0 + (size_t)48 * DIMC;

    v8f acc[2][4];
    #pragma unroll
    for (int i = 0; i < 2; ++i)
        #pragma unroll
        for (int j = 0; j < 4; ++j) acc[i][j] = (v8f)(0.0f);

    for (int k0 = 0; k0 < DIMC; k0 += 32) {
        __builtin_prefetch(b0 + k0 + 512, 0, 1);   // global_prefetch_b8
        __builtin_prefetch(b1 + k0 + 512, 0, 1);
        __builtin_prefetch(b2 + k0 + 512, 0, 1);
        __builtin_prefetch(b3 + k0 + 512, 0, 1);
        v16bf A0 = ldfrag_bf16(a0 + k0);
        v16bf A1 = ldfrag_bf16(a1 + k0);
        v16bf B0 = ldfrag_w(b0 + k0);
        v16bf B1 = ldfrag_w(b1 + k0);
        v16bf B2 = ldfrag_w(b2 + k0);
        v16bf B3 = ldfrag_w(b3 + k0);
        acc[0][0] = WMMA_BF16(A0, B0, acc[0][0]);
        acc[0][1] = WMMA_BF16(A0, B1, acc[0][1]);
        acc[0][2] = WMMA_BF16(A0, B2, acc[0][2]);
        acc[0][3] = WMMA_BF16(A0, B3, acc[0][3]);
        acc[1][0] = WMMA_BF16(A1, B0, acc[1][0]);
        acc[1][1] = WMMA_BF16(A1, B1, acc[1][1]);
        acc[1][2] = WMMA_BF16(A1, B2, acc[1][2]);
        acc[1][3] = WMMA_BF16(A1, B3, acc[1][3]);
    }

    // epilogue: D tile layout -> lane holds (m = r + 8*(lane>=16), n = lane%16)
    #pragma unroll
    for (int i = 0; i < 2; ++i) {
        #pragma unroll
        for (int j = 0; j < 4; ++j) {
            const int col = n0 + j * 16 + lh;
            const int rowb = m0 + i * 16 + ((lane >> 4) << 3);
            const float bc = bias[col];
            if (is_qkv) {
                #pragma unroll
                for (int r = 0; r < 8; ++r)
                    qkv[(size_t)(rowb + r) * QKVC + col] = (__bf16)(acc[i][j][r] + bc);
            } else {
                #pragma unroll
                for (int r = 0; r < 8; ++r)
                    mlp[(size_t)(rowb + r) * MLPC + (col - QKVC)] =
                        (__bf16)gelu_tanh(acc[i][j][r] + bc);
            }
        }
    }
}

// ============== kernel 3: RMSNorm + RoPE on q,k ==============
__global__ __launch_bounds__(256) void rope_kernel(
    const __bf16* __restrict__ qkv, const float* __restrict__ cosb,
    const float* __restrict__ sinb, const float* __restrict__ rqw,
    const float* __restrict__ rkw, __bf16* __restrict__ qr,
    __bf16* __restrict__ kr)
{
    const int s = blockIdx.x;
    const int which = blockIdx.y;                 // 0 = q, 1 = k
    const int lane = threadIdx.x & 31;
    const int wid  = threadIdx.x >> 5;
    const float* rw = which ? rkw : rqw;
    const __bf16* src = qkv + (size_t)s * QKVC + which * DIMC;
    __bf16* dst = (which ? kr : qr) + (size_t)s * DIMC;
    const int d = lane * 4;
    const float c0 = cosb[s * HDC + d + 0], s0 = sinb[s * HDC + d + 0];
    const float c1 = cosb[s * HDC + d + 1], s1 = sinb[s * HDC + d + 1];
    const float c2 = cosb[s * HDC + d + 2], s2 = sinb[s * HDC + d + 2];
    const float c3 = cosb[s * HDC + d + 3], s3 = sinb[s * HDC + d + 3];
    const float w0 = rw[d + 0], w1 = rw[d + 1], w2 = rw[d + 2], w3 = rw[d + 3];

    for (int h = wid; h < HEADSC; h += 8) {
        union { uint2 u; __bf16 e[4]; } xx;
        xx.u = *(const uint2*)&src[h * HDC + d];
        float x0 = (float)xx.e[0], x1 = (float)xx.e[1];
        float x2 = (float)xx.e[2], x3 = (float)xx.e[3];
        float ss = x0*x0 + x1*x1 + x2*x2 + x3*x3;
        #pragma unroll
        for (int off = 16; off > 0; off >>= 1) ss += __shfl_xor(ss, off, 32);
        float rn = rsqrtf(ss * (1.0f / HDC) + EPSC);
        x0 *= rn * w0; x1 *= rn * w1; x2 *= rn * w2; x3 *= rn * w3;
        union { uint2 u; __bf16 e[4]; } yy;
        yy.e[0] = (__bf16)(x0 * c0 - x1 * s0);
        yy.e[1] = (__bf16)(x1 * c1 + x0 * s1);
        yy.e[2] = (__bf16)(x2 * c2 - x3 * s2);
        yy.e[3] = (__bf16)(x3 * c3 + x2 * s3);
        *(uint2*)&dst[h * HDC + d] = yy.u;
    }
}

// ============== kernel 4: flash attention (one head per block) ==============
// 128 thr = 4 waves, each wave owns 16 query rows; 64-key K/V tiles in LDS.
// K tile staged by the Tensor Data Mover (TDM) when available; V^T manually.
__global__ __launch_bounds__(128) void attn_kernel(
    const __bf16* __restrict__ qr, const __bf16* __restrict__ kr,
    const __bf16* __restrict__ qkv, __bf16* __restrict__ attno)
{
    __shared__ __attribute__((aligned(16))) __bf16 kt[64 * HDC];    // [key][d]
    __shared__ __attribute__((aligned(16))) __bf16 vt[HDC * 64];    // [d][key]
    __shared__ __attribute__((aligned(16))) __bf16 pbuf[4][16 * 64];

    const int lane = threadIdx.x & 31;
    const int wid  = threadIdx.x >> 5;
    const int h  = blockIdx.y;
    const int q0 = blockIdx.x * 64 + wid * 16;
    const int lh = lane & 15;
    const int hoff = (lane & 16) ? 8 : 0;
    // softmax in base-2 domain: fold 1/sqrt(128) * log2(e) into one scale
    const float sc2 = 0.08838834764831845f * 1.4426950408889634f;

    v16bf Qf[4];
    {
        const __bf16* qrow = qr + (size_t)(q0 + lh) * DIMC + h * HDC + hoff;
        #pragma unroll
        for (int kk = 0; kk < 4; ++kk) Qf[kk] = ldfrag_bf16(qrow + kk * 32);
    }

    v8f O[8];
    #pragma unroll
    for (int t = 0; t < 8; ++t) O[t] = (v8f)(0.0f);
    float mrow[8], lrow[8];
    #pragma unroll
    for (int r = 0; r < 8; ++r) { mrow[r] = -3.0e38f; lrow[r] = 0.0f; }

    for (int key0 = 0; key0 < SEQC; key0 += 64) {
        __syncthreads();   // protect LDS tiles from previous iteration's readers
#if USE_TDM
        // ---- K tile via Tensor Data Mover (issued once, by wave 0) ----
        if (wid == 0) {
            tdm_load_k_tile((unsigned)(size_t)&kt[0],
                (unsigned long long)(const void*)(kr + (size_t)key0 * DIMC + h * HDC));
        }
        // ---- V^T tile staged manually (TDM cannot transpose) ----
        #pragma unroll
        for (int j = 0; j < 8; ++j) {
            int c = threadIdx.x + 128 * j;
            int key = c >> 4;
            int d   = (c & 15) * 8;
            union { uint4 u; __bf16 e[8]; } vv;
            vv.u = *(const uint4*)&qkv[(size_t)(key0 + key) * QKVC + 2 * DIMC + h * HDC + d];
            #pragma unroll
            for (int e = 0; e < 8; ++e) vt[(d + e) * 64 + key] = vv.e[e];
        }
#if __has_builtin(__builtin_amdgcn_s_wait_tensorcnt)
        if (wid == 0) __builtin_amdgcn_s_wait_tensorcnt(0);
#endif
#else
        // ---- cooperative staging of K tile and V^T tile (64 keys x 128 d) ----
        #pragma unroll
        for (int j = 0; j < 8; ++j) {
            int c = threadIdx.x + 128 * j;          // 0..1023 chunks of 8 elems
            int key = c >> 4;                        // 0..63
            int d   = (c & 15) * 8;
            *(uint4*)&kt[key * HDC + d] =
                *(const uint4*)&kr[(size_t)(key0 + key) * DIMC + h * HDC + d];
            union { uint4 u; __bf16 e[8]; } vv;
            vv.u = *(const uint4*)&qkv[(size_t)(key0 + key) * QKVC + 2 * DIMC + h * HDC + d];
            #pragma unroll
            for (int e = 0; e < 8; ++e) vt[(d + e) * 64 + key] = vv.e[e];
        }
#endif
        __syncthreads();

        // ---- scores: S = Q @ K^T  (four 16-key subtiles) ----
        v8f sc4[4];
        #pragma unroll
        for (int ss = 0; ss < 4; ++ss) sc4[ss] = (v8f)(0.0f);
        #pragma unroll
        for (int kk = 0; kk < 4; ++kk) {
            #pragma unroll
            for (int ss = 0; ss < 4; ++ss) {
                v16bf Bk = ldfrag_bf16(&kt[(ss * 16 + lh) * HDC + kk * 32 + hoff]);
                sc4[ss] = WMMA_BF16(Qf[kk], Bk, sc4[ss]);
            }
        }

        // ---- online softmax (per-row stats across 16-lane groups) ----
        float cf[8];
        #pragma unroll
        for (int r = 0; r < 8; ++r) {
            float a0 = sc4[0][r] * sc2, a1 = sc4[1][r] * sc2;
            float a2 = sc4[2][r] * sc2, a3 = sc4[3][r] * sc2;
            float tm = fmaxf(fmaxf(a0, a1), fmaxf(a2, a3));
            #pragma unroll
            for (int off = 8; off > 0; off >>= 1) tm = fmaxf(tm, __shfl_xor(tm, off, 16));
            float mn = fmaxf(mrow[r], tm);
            float cc = exp2f(mrow[r] - mn);
            float p0 = exp2f(a0 - mn), p1 = exp2f(a1 - mn);
            float p2 = exp2f(a2 - mn), p3 = exp2f(a3 - mn);
            float rs = (p0 + p1) + (p2 + p3);
            #pragma unroll
            for (int off = 8; off > 0; off >>= 1) rs += __shfl_xor(rs, off, 16);
            lrow[r] = lrow[r] * cc + rs;
            mrow[r] = mn;
            cf[r] = cc;
            int m = r + ((lane >> 4) << 3);
            pbuf[wid][m * 64 +  0 + lh] = (__bf16)p0;
            pbuf[wid][m * 64 + 16 + lh] = (__bf16)p1;
            pbuf[wid][m * 64 + 32 + lh] = (__bf16)p2;
            pbuf[wid][m * 64 + 48 + lh] = (__bf16)p3;
        }
        #pragma unroll
        for (int t = 0; t < 8; ++t)
            #pragma unroll
            for (int r = 0; r < 8; ++r) O[t][r] = O[t][r] * cf[r];
        __syncthreads();   // make P visible (cross-lane through LDS)

        // ---- O += P @ V  (A = P from LDS, B = V^T tile) ----
        v16bf Pf0 = ldfrag_bf16(&pbuf[wid][lh * 64 + hoff]);
        v16bf Pf1 = ldfrag_bf16(&pbuf[wid][lh * 64 + 32 + hoff]);
        #pragma unroll
        for (int t = 0; t < 8; ++t) {
            v16bf Vf0 = ldfrag_bf16(&vt[(t * 16 + lh) * 64 + hoff]);
            v16bf Vf1 = ldfrag_bf16(&vt[(t * 16 + lh) * 64 + 32 + hoff]);
            O[t] = WMMA_BF16(Pf0, Vf0, O[t]);
            O[t] = WMMA_BF16(Pf1, Vf1, O[t]);
        }
    }

    // ---- normalize and store ----
    #pragma unroll
    for (int r = 0; r < 8; ++r) {
        float inv = 1.0f / lrow[r];
        int row = q0 + r + ((lane >> 4) << 3);
        #pragma unroll
        for (int t = 0; t < 8; ++t) {
            attno[(size_t)row * DIMC + h * HDC + t * 16 + lh] = (__bf16)(O[t][r] * inv);
        }
    }
}

// ============== kernel 5: GEMM2  cat(attn,mlp) @ proj_w^T + gate/residual ==============
template <typename WT>
__global__ __launch_bounds__(256) void gemm2_kernel(
    const __bf16* __restrict__ attno, const __bf16* __restrict__ mlp,
    const WT* __restrict__ pw, const float* __restrict__ pb,
    const float* __restrict__ mod, const float* __restrict__ hid,
    float* __restrict__ out)
{
    const int lane = threadIdx.x & 31;
    const int wid  = threadIdx.x >> 5;
    const int wm = wid & 3, wn = wid >> 2;
    const int m0 = blockIdx.y * 128 + wm * 32;
    const int n0 = blockIdx.x * 128 + wn * 64;
    const int lh = lane & 15;
    const int hoff = (lane & 16) ? 8 : 0;

    const WT* b0 = pw + (size_t)(n0 + lh) * CATC + hoff;
    const WT* b1 = b0 + (size_t)16 * CATC;
    const WT* b2 = b0 + (size_t)32 * CATC;
    const WT* b3 = b0 + (size_t)48 * CATC;

    v8f acc[2][4];
    #pragma unroll
    for (int i = 0; i < 2; ++i)
        #pragma unroll
        for (int j = 0; j < 4; ++j) acc[i][j] = (v8f)(0.0f);

    // ---- K segment 1: attention output (ld = 3072) ----
    {
        const __bf16* a0 = attno + (size_t)(m0 + lh) * DIMC + hoff;
        const __bf16* a1 = attno + (size_t)(m0 + 16 + lh) * DIMC + hoff;
        for (int k0 = 0; k0 < DIMC; k0 += 32) {
            __builtin_prefetch(b0 + k0 + 512, 0, 1);
            __builtin_prefetch(b1 + k0 + 512, 0, 1);
            __builtin_prefetch(b2 + k0 + 512, 0, 1);
            __builtin_prefetch(b3 + k0 + 512, 0, 1);
            v16bf A0 = ldfrag_bf16(a0 + k0);
            v16bf A1 = ldfrag_bf16(a1 + k0);
            v16bf B0 = ldfrag_w(b0 + k0);
            v16bf B1 = ldfrag_w(b1 + k0);
            v16bf B2 = ldfrag_w(b2 + k0);
            v16bf B3 = ldfrag_w(b3 + k0);
            acc[0][0] = WMMA_BF16(A0, B0, acc[0][0]);
            acc[0][1] = WMMA_BF16(A0, B1, acc[0][1]);
            acc[0][2] = WMMA_BF16(A0, B2, acc[0][2]);
            acc[0][3] = WMMA_BF16(A0, B3, acc[0][3]);
            acc[1][0] = WMMA_BF16(A1, B0, acc[1][0]);
            acc[1][1] = WMMA_BF16(A1, B1, acc[1][1]);
            acc[1][2] = WMMA_BF16(A1, B2, acc[1][2]);
            acc[1][3] = WMMA_BF16(A1, B3, acc[1][3]);
        }
    }
    // ---- K segment 2: gelu(mlp) (ld = 12288), weight cols offset by 3072 ----
    {
        const __bf16* a0 = mlp + (size_t)(m0 + lh) * MLPC + hoff;
        const __bf16* a1 = mlp + (size_t)(m0 + 16 + lh) * MLPC + hoff;
        for (int k0 = 0; k0 < MLPC; k0 += 32) {
            __builtin_prefetch(b0 + DIMC + k0 + 512, 0, 1);
            __builtin_prefetch(b1 + DIMC + k0 + 512, 0, 1);
            __builtin_prefetch(b2 + DIMC + k0 + 512, 0, 1);
            __builtin_prefetch(b3 + DIMC + k0 + 512, 0, 1);
            v16bf A0 = ldfrag_bf16(a0 + k0);
            v16bf A1 = ldfrag_bf16(a1 + k0);
            v16bf B0 = ldfrag_w(b0 + DIMC + k0);
            v16bf B1 = ldfrag_w(b1 + DIMC + k0);
            v16bf B2 = ldfrag_w(b2 + DIMC + k0);
            v16bf B3 = ldfrag_w(b3 + DIMC + k0);
            acc[0][0] = WMMA_BF16(A0, B0, acc[0][0]);
            acc[0][1] = WMMA_BF16(A0, B1, acc[0][1]);
            acc[0][2] = WMMA_BF16(A0, B2, acc[0][2]);
            acc[0][3] = WMMA_BF16(A0, B3, acc[0][3]);
            acc[1][0] = WMMA_BF16(A1, B0, acc[1][0]);
            acc[1][1] = WMMA_BF16(A1, B1, acc[1][1]);
            acc[1][2] = WMMA_BF16(A1, B2, acc[1][2]);
            acc[1][3] = WMMA_BF16(A1, B3, acc[1][3]);
        }
    }

    const float* gate = mod + 2 * DIMC;
    #pragma unroll
    for (int i = 0; i < 2; ++i) {
        #pragma unroll
        for (int j = 0; j < 4; ++j) {
            const int col = n0 + j * 16 + lh;
            const int rowb = m0 + i * 16 + ((lane >> 4) << 3);
            const float bc = pb[col];
            const float g  = gate[col];
            #pragma unroll
            for (int r = 0; r < 8; ++r) {
                int row = rowb + r;
                float vv = acc[i][j][r] + bc;
                out[(size_t)row * DIMC + col] = g * vv + hid[(size_t)row * DIMC + col];
            }
        }
    }
}

// ================================ launch ================================
extern "C" void kernel_launch(void* const* d_in, const int* in_sizes, int n_in,
                              void* d_out, int out_size, void* d_ws, size_t ws_size,
                              hipStream_t stream) {
    const float* hid   = (const float*)d_in[0];
    const float* temb  = (const float*)d_in[1];
    const float* rcos  = (const float*)d_in[2];
    const float* rsin  = (const float*)d_in[3];
    const float* normw = (const float*)d_in[4];
    const float* normb = (const float*)d_in[5];
    const float* qkvw  = (const float*)d_in[6];
    const float* qkvb  = (const float*)d_in[7];
    const float* projw = (const float*)d_in[8];
    const float* projb = (const float*)d_in[9];
    const float* rqw   = (const float*)d_in[10];
    const float* rkw   = (const float*)d_in[11];

    char* ws = (char*)d_ws;
    size_t off = 0;
    auto carve = [&](size_t bytes) {
        char* p = ws + off;
        off += (bytes + 255) & ~(size_t)255;
        return p;
    };
    float*  mod   = (float*) carve((size_t)QKVC * 4);
    __bf16* xn    = (__bf16*)carve((size_t)SEQC * DIMC * 2);
    __bf16* qkv   = (__bf16*)carve((size_t)SEQC * QKVC * 2);
    __bf16* mlp   = (__bf16*)carve((size_t)SEQC * MLPC * 2);
    __bf16* qr    = (__bf16*)carve((size_t)SEQC * DIMC * 2);
    __bf16* kr    = (__bf16*)carve((size_t)SEQC * DIMC * 2);
    __bf16* attno = (__bf16*)carve((size_t)SEQC * DIMC * 2);
    // optional bf16 weight copies (L2-resident: 132 MB + 94.4 MB < 192 MB L2 each)
    __bf16* w1b   = (__bf16*)carve((size_t)N1C * DIMC * 2);
    __bf16* w2b   = (__bf16*)carve((size_t)DIMC * CATC * 2);
    const bool wbf16 = (off <= ws_size);

    mod_gemv_kernel<<<QKVC / 8, 256, 0, stream>>>(temb, normw, normb, mod);
    ln_kernel<<<SEQC, 256, 0, stream>>>(hid, mod, xn);

    if (wbf16) {
        cvt_bf16_kernel<<<((size_t)N1C  * DIMC) / 2048, 256, 0, stream>>>(qkvw,  w1b);
        cvt_bf16_kernel<<<((size_t)DIMC * CATC) / 2048, 256, 0, stream>>>(projw, w2b);
        gemm1_kernel<__bf16><<<dim3(N1C / 128, SEQC / 128), 256, 0, stream>>>(
            xn, w1b, qkvb, qkv, mlp);
    } else {
        gemm1_kernel<float><<<dim3(N1C / 128, SEQC / 128), 256, 0, stream>>>(
            xn, qkvw, qkvb, qkv, mlp);
    }

    rope_kernel<<<dim3(SEQC, 2), 256, 0, stream>>>(qkv, rcos, rsin, rqw, rkw, qr, kr);
    attn_kernel<<<dim3(SEQC / 64, HEADSC), 128, 0, stream>>>(qr, kr, qkv, attno);

    if (wbf16) {
        gemm2_kernel<__bf16><<<dim3(DIMC / 128, SEQC / 128), 256, 0, stream>>>(
            attno, mlp, w2b, projb, mod, hid, (float*)d_out);
    } else {
        gemm2_kernel<float><<<dim3(DIMC / 128, SEQC / 128), 256, 0, stream>>>(
            attno, mlp, projw, projb, mod, hid, (float*)d_out);
    }
}
